// ControllerLSTM_41334765256843
// MI455X (gfx1250) — matrix operand order repaired and checked by reading, not compiled, not run on hardware
//
#include <hip/hip_runtime.h>
#include <hip/hip_bf16.h>
#include <math.h>

// Problem dims
#define HID   2048
#define BATCH 1024
#define NBLK  12
#define FOURH 8192   // 4*HID
#define K2    4096   // concatenated K = 2*HID  ([x | h] vs [W_ih | W_hh])
#define KITER 128    // K2 / 32

#define USE_ASYNC 1  // 1: global_load_async_to_lds_b128 staging; 0: load+ds_store

typedef __attribute__((ext_vector_type(16))) __bf16 v16bf;
typedef __attribute__((ext_vector_type(8)))  __bf16 v8bf;
typedef __attribute__((ext_vector_type(8)))  float  v8f;

// ---- fp32 -> bf16 round-to-nearest-even (bit-level) ----
static __device__ __forceinline__ __bf16 f2bf(float f) {
  union { float f; unsigned u; } in; in.f = f;
  unsigned u = in.u;
  unsigned r = (u + 0x7FFFu + ((u >> 16) & 1u)) >> 16;
  union { unsigned short s; __bf16 b; } out; out.s = (unsigned short)r;
  return out.b;
}

static __device__ __forceinline__ float sigmoidf_(float x) {
  return 1.f / (1.f + __expf(-x));
}
// branch-free tanh: 1 - 2/(1+exp(2x)); saturates correctly at +/-1
static __device__ __forceinline__ float tanhf_(float x) {
  return 1.f - 2.f / (1.f + __expf(2.f * x));
}

// ============================================================================
// Kernel 1: convert + concatenate weights: Wcat[n, 0:2048]=W_ih[n], [2048:4096]=W_hh[n]
// ============================================================================
__global__ __launch_bounds__(256) void convert_weights_kernel(
    const float* __restrict__ Wih, const float* __restrict__ Whh,
    __bf16* __restrict__ Wcat) {
  size_t i4 = ((size_t)blockIdx.x * 256 + threadIdx.x) * 4;
  int n = (int)(i4 >> 12);        // row (0..8191), 4096 per row
  int k = (int)(i4 & 4095);       // col in concatenated K
  const float* src = (k < HID) ? &Wih[(size_t)n * HID + k]
                               : &Whh[(size_t)n * HID + (k - HID)];
  float4 v = *(const float4*)src;
  __bf16* dst = &Wcat[i4];
  dst[0] = f2bf(v.x); dst[1] = f2bf(v.y); dst[2] = f2bf(v.z); dst[3] = f2bf(v.w);
}

// ============================================================================
// Kernel 2: init state: XH=0 (bf16, as u32), c=0, h=0, bias = b_ih + b_hh
// ============================================================================
__global__ __launch_bounds__(256) void init_state_kernel(
    const float* __restrict__ b_ih, const float* __restrict__ b_hh,
    float* __restrict__ bias, float* __restrict__ c, float* __restrict__ h,
    unsigned* __restrict__ XH32) {
  int i = blockIdx.x * 256 + threadIdx.x;
  XH32[i] = 0u;                 // 2M u32 = 4M bf16 = BATCH*K2
  c[i] = 0.f;
  h[i] = 0.f;
  if (i < FOURH) bias[i] = b_ih[i] + b_hh[i];
}

// ============================================================================
// Kernel 3: FUSED WMMA GEMM + LSTM cell, double-buffered async-to-LDS pipeline.
// Block: 256 threads (8 waves). Tile: M=128 rows x J=32 hidden cols across ALL
// FOUR gate regions (n = g*2048 + j). Wave: 2 M-tiles x 4 gates = 8 WMMA/K-step.
// Accumulators start at combined bias. Epilogue is lane-local sigmoid/tanh and
// writes c (f32), h (f32) and h (bf16) into XH[:, 2048:4096].
// ============================================================================
__global__ __launch_bounds__(256) void gemm_lstm_kernel(
    const __bf16* __restrict__ XH, const __bf16* __restrict__ W,
    const float* __restrict__ bias, float* __restrict__ c,
    float* __restrict__ h, __bf16* __restrict__ XHout) {
  // two buffers: [buf][128 rows][40 bf16] (32 k + 8 pad -> 80B row stride)
  __shared__ __align__(16) __bf16 ldsA[2][128 * 40];
  __shared__ __align__(16) __bf16 ldsB[2][128 * 40];

  const int tid  = threadIdx.x;
  const int lane = tid & 31;
  const int wave = tid >> 5;
  const int wm   = wave >> 1;      // 0..3  (32 M-rows each)
  const int wj   = wave & 1;       // 0..1  (16 j-cols each)
  const int lrow = lane & 15;
  const int kh   = lane >> 4;
  const int blockM = blockIdx.y * 128;
  const int blockJ = blockIdx.x * 32;

  // per-thread staging chunk coordinates (512 x 16B chunks each for A and B)
  const int c0row = tid >> 2;            // 0..63
  const int c0seg = tid & 3;             // 0..3
  const int bgrow0 = ((c0row >> 5) * HID) + blockJ + (c0row & 31);
  const int bgrow1 = bgrow0 + 2 * HID;   // lds row + 64 => gate + 2

  const unsigned laA0 = (unsigned)(uintptr_t)&ldsA[0][0] +
                        (unsigned)(c0row * 80 + c0seg * 16);
  const unsigned laB0 = (unsigned)(uintptr_t)&ldsB[0][0] +
                        (unsigned)(c0row * 80 + c0seg * 16);

  const __bf16* gA0 = &XH[(size_t)(blockM + c0row) * K2 + c0seg * 8];
  const __bf16* gB0 = &W[(size_t)bgrow0 * K2 + c0seg * 8];
  const __bf16* gB1 = &W[(size_t)bgrow1 * K2 + c0seg * 8];

#define ISSUE_TILE(it_)                                                        \
  do {                                                                         \
    const int kk_ = (it_) * 32;                                                \
    const unsigned bofs_ = ((it_) & 1) ? 10240u : 0u;                          \
    const __bf16* pa0_ = gA0 + kk_;                                            \
    const __bf16* pa1_ = pa0_ + (size_t)64 * K2;                               \
    const __bf16* pb0_ = gB0 + kk_;                                            \
    const __bf16* pb1_ = gB1 + kk_;                                            \
    asm volatile("global_load_async_to_lds_b128 %0, %1, off"                   \
                 :: "v"(laA0 + bofs_), "v"(pa0_) : "memory");                  \
    asm volatile("global_load_async_to_lds_b128 %0, %1, off"                   \
                 :: "v"(laA0 + bofs_ + 5120u), "v"(pa1_) : "memory");          \
    asm volatile("global_load_async_to_lds_b128 %0, %1, off"                   \
                 :: "v"(laB0 + bofs_), "v"(pb0_) : "memory");                  \
    asm volatile("global_load_async_to_lds_b128 %0, %1, off"                   \
                 :: "v"(laB0 + bofs_ + 5120u), "v"(pb1_) : "memory");          \
  } while (0)

#define ISSUE_TILE_SYNC(it_)                                                   \
  do {                                                                         \
    const int kk_ = (it_) * 32;                                                \
    const int b_ = (it_) & 1;                                                  \
    *(v8bf*)&ldsA[b_][c0row * 40 + c0seg * 8]        = *(const v8bf*)(gA0 + kk_); \
    *(v8bf*)&ldsA[b_][(c0row + 64) * 40 + c0seg * 8] =                         \
        *(const v8bf*)(gA0 + kk_ + (size_t)64 * K2);                           \
    *(v8bf*)&ldsB[b_][c0row * 40 + c0seg * 8]        = *(const v8bf*)(gB0 + kk_); \
    *(v8bf*)&ldsB[b_][(c0row + 64) * 40 + c0seg * 8] = *(const v8bf*)(gB1 + kk_); \
  } while (0)

  // accumulators initialized with combined bias (depends only on column n)
  v8f acc[2][4];
#pragma unroll
  for (int g = 0; g < 4; ++g) {
    float bv = bias[g * HID + blockJ + wj * 16 + lrow];
    v8f bi = {bv, bv, bv, bv, bv, bv, bv, bv};
    acc[0][g] = bi;
    acc[1][g] = bi;
  }

  // prologue: stage tile 0 into buffer 0
#if USE_ASYNC
  ISSUE_TILE(0);
#else
  ISSUE_TILE_SYNC(0);
#endif

  for (int it = 0; it < KITER; ++it) {
    // barrier #1: everyone done READING buf[(it+1)&1] (last read at it-1)
    __syncthreads();
#if USE_ASYNC
    if (it + 1 < KITER) {
      ISSUE_TILE(it + 1);
      asm volatile("s_wait_asynccnt 0x4" ::: "memory");  // tile `it` landed
    } else {
      asm volatile("s_wait_asynccnt 0x0" ::: "memory");
    }
#else
    if (it + 1 < KITER) ISSUE_TILE_SYNC(it + 1);
#endif
    // barrier #2: all waves' tile-`it` data visible in LDS
    __syncthreads();

    const __bf16* baseA = &ldsA[it & 1][0];
    const __bf16* baseB = &ldsB[it & 1][0];

    // A fragments (16x32 bf16, ISA layout)
    v16bf afrag[2];
#pragma unroll
    for (int mt = 0; mt < 2; ++mt) {
      const __bf16* p = baseA + (wm * 32 + mt * 16 + lrow) * 40 + kh * 8;
      v8bf lo = *(const v8bf*)p;
      v8bf hi = *(const v8bf*)(p + 16);
      afrag[mt] = __builtin_shufflevector(lo, hi,
          0, 1, 2, 3, 4, 5, 6, 7, 8, 9, 10, 11, 12, 13, 14, 15);
    }
    // B fragments (32x16): lane = column n, K = (lane/16)*8 +0..7 and 16+(lane/16)*8 +0..7
    v16bf bfrag[4];
#pragma unroll
    for (int g = 0; g < 4; ++g) {
      const __bf16* p = baseB + (g * 32 + wj * 16 + lrow) * 40 + kh * 8;
      v8bf lo = *(const v8bf*)p;
      v8bf hi = *(const v8bf*)(p + 16);
      bfrag[g] = __builtin_shufflevector(lo, hi,
          0, 1, 2, 3, 4, 5, 6, 7, 8, 9, 10, 11, 12, 13, 14, 15);
    }

#pragma unroll
    for (int mt = 0; mt < 2; ++mt)
#pragma unroll
      for (int g = 0; g < 4; ++g)
        acc[mt][g] = __builtin_amdgcn_wmma_f32_16x16x32_bf16(
            false, afrag[mt], false, bfrag[g], (short)0, acc[mt][g],
            false, false);
  }
#undef ISSUE_TILE
#undef ISSUE_TILE_SYNC

  // Epilogue: lane-local LSTM cell. D layout: VGPR jv -> m = (lane/16)*8 + jv,
  // n(=j) = lane%16. All four gates of (m,j) live in this lane.
  const int j = blockJ + wj * 16 + lrow;
#pragma unroll
  for (int mt = 0; mt < 2; ++mt) {
    const int m0 = blockM + wm * 32 + mt * 16 + kh * 8;
#pragma unroll
    for (int jv = 0; jv < 8; ++jv) {
      const size_t off = (size_t)(m0 + jv) * HID + j;
      float gi = acc[mt][0][jv];
      float gf = acc[mt][1][jv];
      float gg = acc[mt][2][jv];
      float go = acc[mt][3][jv];
      float c2 = sigmoidf_(gf) * c[off] + sigmoidf_(gi) * tanhf_(gg);
      float h2 = sigmoidf_(go) * tanhf_(c2);
      c[off] = c2;
      h[off] = h2;
      XHout[(size_t)(m0 + jv) * K2 + HID + j] = f2bf(h2);
    }
  }
}

// ============================================================================
// Kernel 4: decode logits (<=11 classes), argmax, embed next input into XH[:,0:2048].
// One block per batch row.
// ============================================================================
__global__ __launch_bounds__(256) void decode_embed_kernel(
    const float* __restrict__ h, const float* __restrict__ dec, int ncls,
    int* __restrict__ idxbuf, const float* __restrict__ enc,
    __bf16* __restrict__ XH) {
  __shared__ float sred[256];
  __shared__ float slog[16];
  __shared__ int   sidx;
  const int b = blockIdx.x;
  const int tid = threadIdx.x;

  float part[11];
  for (int cl = 0; cl < 11; ++cl) part[cl] = 0.f;
  for (int k = tid; k < HID; k += 256) {
    float hv = h[(size_t)b * HID + k];
    for (int cl = 0; cl < ncls; ++cl)
      part[cl] += hv * dec[(size_t)cl * HID + k];
  }
  for (int cl = 0; cl < ncls; ++cl) {
    sred[tid] = part[cl];
    __syncthreads();
    for (int off = 128; off > 0; off >>= 1) {
      if (tid < off) sred[tid] += sred[tid + off];
      __syncthreads();
    }
    if (tid == 0) slog[cl] = sred[0];
    __syncthreads();
  }
  if (tid == 0) {
    float best = slog[0];
    int bi = 0;
    for (int cl = 1; cl < ncls; ++cl)
      if (slog[cl] > best) { best = slog[cl]; bi = cl; }
    sidx = bi;
    idxbuf[b] = bi;
  }
  __syncthreads();
  if (enc != nullptr) {
    int id = sidx;
    for (int k = tid; k < HID; k += 256)
      XH[(size_t)b * K2 + k] = f2bf(enc[(size_t)id * HID + k]);
  }
}

// ============================================================================
// Kernel 5: pack outputs (idx as float, then h, then c)
// ============================================================================
__global__ __launch_bounds__(256) void finalize_kernel(
    const int* __restrict__ idx, const float* __restrict__ h,
    const float* __restrict__ c, float* __restrict__ out) {
  int i = blockIdx.x * 256 + threadIdx.x;
  const int nh = BATCH * HID;
  if (i < BATCH) {
    out[i] = (float)idx[i];
  } else if (i < BATCH + nh) {
    out[i] = h[i - BATCH];
  } else if (i < BATCH + 2 * nh) {
    out[i] = c[i - BATCH - nh];
  }
}

// ============================================================================
// Host launcher
// ============================================================================
extern "C" void kernel_launch(void* const* d_in, const int* in_sizes, int n_in,
                              void* d_out, int out_size, void* d_ws, size_t ws_size,
                              hipStream_t stream) {
  (void)in_sizes; (void)n_in; (void)out_size; (void)ws_size;
  const float* W_ih      = (const float*)d_in[1];
  const float* W_hh      = (const float*)d_in[2];
  const float* b_ih      = (const float*)d_in[3];
  const float* b_hh      = (const float*)d_in[4];
  const float* enc_act   = (const float*)d_in[5];   // (11, 4, H)
  const float* enc_block = (const float*)d_in[6];   // (11, 11, H)
  const float* dec_act   = (const float*)d_in[7];   // (12, 4, H)
  const float* dec_block = (const float*)d_in[8];   // (11, 11, H)

  char* ws = (char*)d_ws;
  __bf16* Wcat  = (__bf16*)(ws);                     // 8192*4096*2 = 64 MiB
  __bf16* XH    = (__bf16*)(ws + 67108864);          // 1024*4096*2 =  8 MiB
  float*  c     = (float*) (ws + 75497472);          // 1024*2048*4 =  8 MiB
  float*  h     = (float*) (ws + 83886080);          // 1024*2048*4 =  8 MiB
  float*  bias  = (float*) (ws + 92274688);          // 8192*4      = 32 KiB
  int*    idx   = (int*)   (ws + 92307456);          // 1024*4      =  4 KiB

  // One-time per call: weight conversion (fp32 -> bf16, concat K) + state init
  convert_weights_kernel<<<32768, 256, 0, stream>>>(W_ih, W_hh, Wcat);
  init_state_kernel<<<8192, 256, 0, stream>>>(b_ih, b_hh, bias, c, h,
                                              (unsigned*)XH);

  // 23 sequential LSTM steps: fused GEMM+cell, then decode/argmax/embed
  for (int s = 0; s < 23; ++s) {
    gemm_lstm_kernel<<<dim3(64, 8), 256, 0, stream>>>(XH, Wcat, bias, c, h, XH);

    const float* dec;
    const float* enc;
    int ncls;
    if (s == 0) {                       // dec_act[0], next emb = enc_act[0]
      dec = dec_act; ncls = 4; enc = enc_act;
    } else if (s & 1) {                 // bid = t+1: dec_block[bid-1], ncls=bid
      int t = (s - 1) >> 1;
      dec = dec_block + (size_t)t * 11 * HID;
      ncls = t + 1;
      enc = enc_block + (size_t)t * 11 * HID;   // enc_block[bid-1]
    } else {                            // bid = t: dec_act[bid], next emb enc_act[bid]
      int t = s >> 1;
      dec = dec_act + (size_t)t * 4 * HID;
      ncls = 4;
      enc = (s < 22) ? (enc_act + (size_t)t * 4 * HID) : nullptr;
    }
    decode_embed_kernel<<<1024, 256, 0, stream>>>(h, dec, ncls, idx, enc, XH);
  }

  finalize_kernel<<<16388, 256, 0, stream>>>(idx, h, c, (float*)d_out);
}
